// CROSS_SS_MIL_25383256720170
// MI455X (gfx1250) — compile-verified, matrix operand order checked by reading, not run on hardware
//
#include <hip/hip_runtime.h>
#include <math.h>

// ---------------------------------------------------------------------------
// CROSS_SS_MIL for MI455X (gfx1250, wave32). All GEMMs use
// v_wmma_f32_16x16x32_bf16 (fp32->bf16 staged through LDS, fp32 accum).
// The sequential selective scan keeps the 512x128 state in VGPRs and
// double-buffers its per-step operand staging with GLOBAL_LOAD_ASYNC_TO_LDS
// (ASYNCcnt) when the toolchain exposes the builtin.
// ---------------------------------------------------------------------------

typedef __attribute__((ext_vector_type(16))) __bf16 v16bf;
typedef __attribute__((ext_vector_type(8)))  float  v8f;

#define LC 2048
#define DM 512
#define NS 128
#define RK 32

// -------- gfx1250 async global->LDS copy (fallback to sync staging) --------
#if defined(__has_builtin)
#if __has_builtin(__builtin_amdgcn_global_load_async_to_lds_b32) && \
    __has_builtin(__builtin_amdgcn_s_wait_asynccnt)
#define USE_ASYNC_LDS 1
#endif
#endif
#ifndef USE_ASYNC_LDS
#define USE_ASYNC_LDS 0
#endif

#if USE_ASYNC_LDS
// builtin signature (probe-confirmed): (AS1 int*, AS3 int*, imm offset, imm cpol)
#define ASYNC_CP32(gp, lp)                                                   \
  __builtin_amdgcn_global_load_async_to_lds_b32(                             \
      (__attribute__((address_space(1))) int*)(gp),                          \
      (__attribute__((address_space(3))) int*)(lp), 0, 0)
#define ASYNC_WAIT() __builtin_amdgcn_s_wait_asynccnt(0)
#endif

// ------------------------------ GEMM ---------------------------------------
// C[M,N] = A[M,K] @ B[K,N] (+bias, +relu/softplus). A optionally row-flipped.
// Block: 256 threads (8 waves). Tile: BM=128 x BN=64, BK=64 (2 wmma k-steps).
// Wave w -> 32x32 sub-tile: rows wm=(w&3)*32, cols wn=(w>>2)*32, 2x2 frags.
constexpr int BM = 128, BN = 64, BK = 64;
constexpr int LDS_STR = BK + 8;   // 72 bf16 = 144 B rows -> 16B-aligned chunks

__device__ __forceinline__ v16bf load_frag(const __bf16* p) {
  // fragment element i holds k = base + (i<8 ? i : i+8); p points at k=base.
  v16bf r;
#pragma unroll
  for (int i = 0; i < 8; ++i) r[i] = p[i];
#pragma unroll
  for (int i = 0; i < 8; ++i) r[8 + i] = p[16 + i];
  return r;
}

__device__ __forceinline__ float softplus_f(float v) {
  return (v > 20.f) ? v : log1pf(__expf(v));
}

__launch_bounds__(256)
__global__ void gemm_wmma_bf16(const float* __restrict__ A, int lda,
                               const float* __restrict__ B, int ldb,
                               float* __restrict__ C, int ldc,
                               int M, int N, int K,
                               const float* __restrict__ bias,
                               int epi /*0 none,1 bias,2 bias+relu,3 bias+softplus*/,
                               int flipA) {
  __shared__ __bf16 sA[BM * LDS_STR];           // 18 KB
  __shared__ __bf16 sBt[BN * LDS_STR];          //  9 KB (stored [n][k])
  const int tid  = threadIdx.x;
  const int lane = tid & 31;
  const int w    = tid >> 5;
  const int wm   = (w & 3) * 32;
  const int wn   = (w >> 2) * 32;
  const int m0   = blockIdx.y * BM;
  const int n0   = blockIdx.x * BN;
  const int half = lane >> 4;                   // 0: lanes 0-15, 1: 16-31
  const int ml   = lane & 15;
  const int kb   = half * 8;                    // fragment k-chunk base

  v8f acc[2][2];
  const v8f vzero = {0.f,0.f,0.f,0.f,0.f,0.f,0.f,0.f};
#pragma unroll
  for (int i = 0; i < 2; ++i)
#pragma unroll
    for (int j = 0; j < 2; ++j) acc[i][j] = vzero;

  const bool n_interior = (n0 + BN <= N);

  for (int k0 = 0; k0 < K; k0 += BK) {
    { // --- stage A tile: 128 rows x 64 k (each thread: 1 half-row of 32) ---
      // K is always a multiple of 32 here -> whole 32-chunk valid or not.
      const int r  = tid >> 1;
      const int cb = (tid & 1) * 32;
      const int gr = m0 + r;
      const int grow = flipA ? (M - 1 - gr) : gr;
      __bf16* dst = &sA[r * LDS_STR + cb];
      if ((gr < M) && (k0 + cb) < K) {
        const float* src = A + (long)grow * lda + k0 + cb;
#pragma unroll
        for (int c = 0; c < 32; c += 4) {
          const float4 v = *(const float4*)(src + c);
          dst[c]   = (__bf16)v.x; dst[c+1] = (__bf16)v.y;
          dst[c+2] = (__bf16)v.z; dst[c+3] = (__bf16)v.w;
        }
      } else {
#pragma unroll
        for (int c = 0; c < 32; ++c) dst[c] = (__bf16)0.f;
      }
    }
    { // --- stage B tile transposed: sBt[n][k], 64k x 64n -------------------
      const int kk = tid >> 2;                  // 0..63
      const int nb = (tid & 3) * 16;
      const int gk = k0 + kk;
      const float* src = B + (long)gk * ldb + n0 + nb;
      if ((gk < K) && n_interior) {
        float4 v0 = *(const float4*)(src + 0);
        float4 v1 = *(const float4*)(src + 4);
        float4 v2 = *(const float4*)(src + 8);
        float4 v3 = *(const float4*)(src + 12);
        const float vv[16] = {v0.x,v0.y,v0.z,v0.w, v1.x,v1.y,v1.z,v1.w,
                              v2.x,v2.y,v2.z,v2.w, v3.x,v3.y,v3.z,v3.w};
#pragma unroll
        for (int j = 0; j < 16; ++j)
          sBt[(nb + j) * LDS_STR + kk] = (__bf16)vv[j];
      } else if (gk < K) {
#pragma unroll
        for (int j = 0; j < 16; ++j) {
          const int gn = n0 + nb + j;
          const float v = (gn < N) ? src[j] : 0.f;
          sBt[(nb + j) * LDS_STR + kk] = (__bf16)v;
        }
      } else {
#pragma unroll
        for (int j = 0; j < 16; ++j)
          sBt[(nb + j) * LDS_STR + kk] = (__bf16)0.f;
      }
    }
    __syncthreads();

#pragma unroll
    for (int ks = 0; ks < BK; ks += 32) {
      v16bf af[2], bf[2];
#pragma unroll
      for (int i = 0; i < 2; ++i) {
        af[i] = load_frag(&sA [(wm + i*16 + ml) * LDS_STR + ks + kb]);
        bf[i] = load_frag(&sBt[(wn + i*16 + ml) * LDS_STR + ks + kb]);
      }
#pragma unroll
      for (int i = 0; i < 2; ++i)
#pragma unroll
        for (int j = 0; j < 2; ++j)
          acc[i][j] = __builtin_amdgcn_wmma_f32_16x16x32_bf16(
              false, af[i], false, bf[j], (short)0, acc[i][j], false, false);
    }
    __syncthreads();
  }

  // --- epilogue: D-layout VGPR e -> row = e + half*8, col = ml --------------
#pragma unroll
  for (int i = 0; i < 2; ++i)
#pragma unroll
    for (int j = 0; j < 2; ++j)
#pragma unroll
      for (int e = 0; e < 8; ++e) {
        const int row = m0 + wm + i * 16 + half * 8 + e;
        const int col = n0 + wn + j * 16 + ml;
        if (row < M && col < N) {
          float v = acc[i][j][e];
          if (epi >= 1) v += bias[col];
          if (epi == 2) v = fmaxf(v, 0.f);
          if (epi == 3) v = softplus_f(v);
          C[(long)row * ldc + col] = v;
        }
      }
}

// -------------------- depthwise causal conv (K=4) + SiLU -------------------
__global__ void conv_silu_k(const float* __restrict__ xz,   // [T,1024], x = cols 0..511
                            const float* __restrict__ cw,   // [512,4]
                            const float* __restrict__ cb,   // [512]
                            float* __restrict__ xc, int T) {
  const int idx = blockIdx.x * blockDim.x + threadIdx.x;
  if (idx >= T * DM) return;
  const int t = idx >> 9, d = idx & (DM - 1);
  float a = cb[d];
#pragma unroll
  for (int k = 0; k < 4; ++k) {
    const int tt = t - 3 + k;
    if (tt >= 0) a += xz[(long)tt * 1024 + d] * cw[d * 4 + k];
  }
  xc[idx] = a / (1.f + __expf(-a));             // silu
}

// ----------------------------- selective scan ------------------------------
// grid = 8 blocks x 256 thr; thread = (channel d, n-quarter q): 32 states in
// registers.  Per step: double-buffered async staging of B/C/dt/x into LDS,
// 32 exp+fma, 4-lane butterfly reduction over n, fused (y + x*D)*silu(z).
// Stage buffer layout: [0,128)=B, [128,256)=C, [256,320)=dt, [320,384)=x.
__launch_bounds__(256)
__global__ void scan_k(const float* __restrict__ dt,     // [T,512] (softplus'ed)
                       const float* __restrict__ xdbl,   // [T,288]  B@+32 C@+160
                       const float* __restrict__ xconv,  // [T,512]
                       const float* __restrict__ xz,     // [T,1024] z = cols 512..
                       const float* __restrict__ A_log,  // [512,128]
                       const float* __restrict__ Dp,     // [512]
                       float* __restrict__ y, int T) {
  __shared__ float stg[2][384];
  const int tid  = threadIdx.x;
  const int dl   = tid >> 2;                    // 0..63
  const int q    = tid & 3;                     // n-quarter
  const int d    = blockIdx.x * 64 + dl;
  const int base = blockIdx.x * 64;
  float Av[32], h[32];
#pragma unroll
  for (int j = 0; j < 32; ++j) {
    Av[j] = -__expf(A_log[d * NS + q * 32 + j]);
    h[j]  = 0.f;
  }
  const float Dv = Dp[d];

#if USE_ASYNC_LDS
  { // prologue: async-stage step 0
    const float* bc = xdbl;
    float* buf = stg[0];
    if (tid < 128)      ASYNC_CP32(bc + 32 + tid,          buf + tid);
    else                ASYNC_CP32(bc + 160 + (tid - 128), buf + tid);
    if (tid < 64)       ASYNC_CP32(dt + base + tid,        buf + 256 + tid);
    else if (tid < 128) ASYNC_CP32(xconv + base + (tid-64), buf + 320 + (tid-64));
    ASYNC_WAIT();
    __syncthreads();
  }
#endif

  for (int t = 0; t < T; ++t) {
    float* cur = stg[t & 1];
#if USE_ASYNC_LDS
    if (t + 1 < T) {  // overlap next step's staging with this step's math
      const float* bc = xdbl + (long)(t + 1) * 288;
      float* nxt = stg[(t + 1) & 1];
      if (tid < 128)      ASYNC_CP32(bc + 32 + tid,          nxt + tid);
      else                ASYNC_CP32(bc + 160 + (tid - 128), nxt + tid);
      if (tid < 64)       ASYNC_CP32(dt + (long)(t+1) * DM + base + tid,
                                     nxt + 256 + tid);
      else if (tid < 128) ASYNC_CP32(xconv + (long)(t+1) * DM + base + (tid-64),
                                     nxt + 320 + (tid-64));
    }
#else
    { // synchronous staging fallback
      const float* bc = xdbl + (long)t * 288;
      if (tid < 128)      cur[tid] = bc[32 + tid];
      else                cur[tid] = bc[160 + (tid - 128)];
      if (tid < 64)       cur[256 + tid]        = dt   [(long)t * DM + base + tid];
      else if (tid < 128) cur[320 + (tid - 64)] = xconv[(long)t * DM + base + (tid-64)];
      __syncthreads();
    }
#endif
    if (tid == 0 && t + 2 < T) __builtin_prefetch(xdbl + (long)(t + 2) * 288, 0, 1);

    const float dtv = cur[256 + dl], xv = cur[320 + dl], dtx = dtv * xv;
    float part = 0.f;
#pragma unroll
    for (int j = 0; j < 32; ++j) {
      const float dA = __expf(dtv * Av[j]);
      h[j] = h[j] * dA + dtx * cur[q * 32 + j];
      part += h[j] * cur[128 + q * 32 + j];
    }
    part += __shfl_xor(part, 1, 32);            // reduce the 4 n-quarters
    part += __shfl_xor(part, 2, 32);
    if (q == 0) {
      const float zv = xz[(long)t * 1024 + DM + d];
      const float sz = zv / (1.f + __expf(-zv));
      y[(long)t * DM + d] = (part + xv * Dv) * sz;
    }
#if USE_ASYNC_LDS
    ASYNC_WAIT();                               // next-step buffer landed
    __syncthreads();
#else
    __syncthreads();
#endif
  }
}

// -------------------------- row softmax (cols=512) -------------------------
__launch_bounds__(256)
__global__ void softmax_rows_k(const float* __restrict__ in, float* __restrict__ out,
                               int cols) {
  const int t = blockIdx.x, tid = threadIdx.x;
  const float* r = in + (long)t * cols;
  __shared__ float red[8];
  float m = -3.4e38f;
  for (int i = tid; i < cols; i += 256) m = fmaxf(m, r[i]);
  for (int s = 16; s; s >>= 1) m = fmaxf(m, __shfl_xor(m, s, 32));
  if ((tid & 31) == 0) red[tid >> 5] = m;
  __syncthreads();
  if (tid == 0) { float v = red[0]; for (int i = 1; i < 8; ++i) v = fmaxf(v, red[i]); red[0] = v; }
  __syncthreads();
  m = red[0];
  __syncthreads();
  float s_ = 0.f;
  for (int i = tid; i < cols; i += 256) s_ += __expf(r[i] - m);
  for (int s = 16; s; s >>= 1) s_ += __shfl_xor(s_, s, 32);
  if ((tid & 31) == 0) red[tid >> 5] = s_;
  __syncthreads();
  if (tid == 0) { float v = 0.f; for (int i = 0; i < 8; ++i) v += red[i]; red[0] = v; }
  __syncthreads();
  const float inv = 1.f / red[0];
  for (int i = tid; i < cols; i += 256) out[(long)t * cols + i] = __expf(r[i] - m) * inv;
}

// ------------- build seq_fi / seq_if (concat + flip + cls token) -----------
__global__ void assemble_k(const float* __restrict__ yf, const float* __restrict__ yi_rev,
                           const float* __restrict__ cls1, const float* __restrict__ cls2,
                           float* __restrict__ sfi, float* __restrict__ sif) {
  const int idx = blockIdx.x * blockDim.x + threadIdx.x;
  if (idx >= (LC + 1) * 1024) return;
  const int t = idx >> 10, c = idx & 1023;
  float vfi, vif;
  if (t == LC) { vfi = cls1[c]; vif = cls2[c]; }
  else {
    const int hf = c >> 9, d = c & 511;
    const float a = yf[(long)t * DM + d];
    const float b = yi_rev[(long)(LC - 1 - t) * DM + d];   // flip
    vfi = hf ? b : a;
    vif = hf ? a : b;
  }
  sfi[idx] = vfi;
  sif[idx] = vif;
}

__global__ void copy_row_k(const float* __restrict__ src, float* __restrict__ dst) {
  const int i = blockIdx.x * blockDim.x + threadIdx.x;
  if (i < DM) dst[i] = src[i];
}

__launch_bounds__(256)
__global__ void classifier_k(const float* __restrict__ last /*[1024]*/,
                             const float* __restrict__ w /*[1024,2]*/,
                             const float* __restrict__ b, float* __restrict__ out) {
  __shared__ float r0[256], r1[256];
  float s0 = 0.f, s1 = 0.f;
  for (int i = threadIdx.x; i < 1024; i += 256) {
    const float v = last[i];
    s0 += v * w[i * 2 + 0];
    s1 += v * w[i * 2 + 1];
  }
  r0[threadIdx.x] = s0; r1[threadIdx.x] = s1;
  __syncthreads();
  for (int s = 128; s > 0; s >>= 1) {
    if (threadIdx.x < s) { r0[threadIdx.x] += r0[threadIdx.x + s]; r1[threadIdx.x] += r1[threadIdx.x + s]; }
    __syncthreads();
  }
  if (threadIdx.x == 0) { out[0] = r0[0] + b[0]; out[1] = r1[0] + b[1]; }
}

// --------------------------------- host ------------------------------------
static inline int cdiv(int a, int b) { return (a + b - 1) / b; }

// workspace layout (floats)
static const long SZ_F   = 2048L * 512;
static const long SZ_XZ  = 2049L * 1024;
static const long SZ_XC  = 2049L * 512;
static const long SZ_XD  = 2049L * 288;
static const long SZ_D   = 2049L * 512;

static const long OFF_F    = 0;
static const long OFF_XZ   = OFF_F    + SZ_F;
static const long OFF_XC   = OFF_XZ   + SZ_XZ;
static const long OFF_XD   = OFF_XC   + SZ_XC;
static const long OFF_DT   = OFF_XD   + SZ_XD;
static const long OFF_Y    = OFF_DT   + SZ_D;
static const long OFF_OUT  = OFF_Y    + SZ_D;
static const long OFF_YF   = OFF_OUT  + SZ_D;
static const long OFF_YI   = OFF_YF   + SZ_F;
static const long OFF_SFI  = OFF_YI   + SZ_F;
static const long OFF_SIF  = OFF_SFI  + SZ_XZ;
static const long OFF_LAST = OFF_SIF  + SZ_XZ;   // [1024]: of_last, oi_last

static void run_mamba(float* ws, const float* inA, int lda, int T, int flip,
                      const float* in_w, int Kin,
                      const float* conv_w, const float* conv_b,
                      const float* xproj_w, const float* dt_w, const float* dt_b,
                      const float* A_log, const float* Dp, const float* out_w,
                      float* outbuf, hipStream_t stream) {
  // in_proj -> xz [T,1024]
  gemm_wmma_bf16<<<dim3(cdiv(1024, BN), cdiv(T, BM)), 256, 0, stream>>>(
      inA, lda, in_w, 1024, ws + OFF_XZ, 1024, T, 1024, Kin, nullptr, 0, flip);
  // causal depthwise conv + silu -> xconv [T,512]
  conv_silu_k<<<cdiv(T * DM, 256), 256, 0, stream>>>(ws + OFF_XZ, conv_w, conv_b,
                                                     ws + OFF_XC, T);
  // x_proj -> x_dbl [T,288]
  gemm_wmma_bf16<<<dim3(cdiv(288, BN), cdiv(T, BM)), 256, 0, stream>>>(
      ws + OFF_XC, DM, xproj_w, 288, ws + OFF_XD, 288, T, 288, DM, nullptr, 0, 0);
  // dt = softplus(dt_in @ dt_w + dt_b) -> [T,512]
  gemm_wmma_bf16<<<dim3(cdiv(DM, BN), cdiv(T, BM)), 256, 0, stream>>>(
      ws + OFF_XD, 288, dt_w, DM, ws + OFF_DT, DM, T, DM, RK, dt_b, 3, 0);
  // selective scan (fused +x*D and silu(z) gate) -> y [T,512]
  scan_k<<<8, 256, 0, stream>>>(ws + OFF_DT, ws + OFF_XD, ws + OFF_XC,
                                ws + OFF_XZ, A_log, Dp, ws + OFF_Y, T);
  // out_proj
  gemm_wmma_bf16<<<dim3(cdiv(DM, BN), cdiv(T, BM)), 256, 0, stream>>>(
      ws + OFF_Y, DM, out_w, DM, outbuf, DM, T, DM, DM, nullptr, 0, 0);
}

extern "C" void kernel_launch(void* const* d_in, const int* in_sizes, int n_in,
                              void* d_out, int out_size, void* d_ws, size_t ws_size,
                              hipStream_t stream) {
  const float* x      = (const float*)d_in[0];
  const float* feat_w = (const float*)d_in[1];
  const float* feat_b = (const float*)d_in[2];
  const float* cls1   = (const float*)d_in[3];
  const float* cls2   = (const float*)d_in[4];
  const float* cls_w  = (const float*)d_in[5];
  const float* cls_b  = (const float*)d_in[6];
  const float* m_in_w   = (const float*)d_in[7];
  const float* m_conv_w = (const float*)d_in[8];
  const float* m_conv_b = (const float*)d_in[9];
  const float* m_xproj  = (const float*)d_in[10];
  const float* m_dt_w   = (const float*)d_in[11];
  const float* m_dt_b   = (const float*)d_in[12];
  const float* m_A_log  = (const float*)d_in[13];
  const float* m_D      = (const float*)d_in[14];
  const float* m_out_w  = (const float*)d_in[15];
  const float* c_in_w   = (const float*)d_in[16];
  const float* c_conv_w = (const float*)d_in[17];
  const float* c_conv_b = (const float*)d_in[18];
  const float* c_xproj  = (const float*)d_in[19];
  const float* c_dt_w   = (const float*)d_in[20];
  const float* c_dt_b   = (const float*)d_in[21];
  const float* c_A_log  = (const float*)d_in[22];
  const float* c_D      = (const float*)d_in[23];
  const float* c_out_w  = (const float*)d_in[24];

  float* ws  = (float*)d_ws;
  float* out = (float*)d_out;

  // features: f = relu(x @ feat_w + feat_b)  [2048,512]
  gemm_wmma_bf16<<<dim3(cdiv(DM, BN), cdiv(LC, BM)), 256, 0, stream>>>(
      x, 1024, feat_w, DM, ws + OFF_F, DM, LC, DM, 1024, feat_b, 2, 0);

  // level 1, forward
  run_mamba(ws, ws + OFF_F, DM, LC, 0, m_in_w, DM, m_conv_w, m_conv_b, m_xproj,
            m_dt_w, m_dt_b, m_A_log, m_D, m_out_w, ws + OFF_OUT, stream);
  softmax_rows_k<<<LC, 256, 0, stream>>>(ws + OFF_OUT, ws + OFF_YF, DM);

  // level 1, reverse (A rows flipped inside in_proj GEMM)
  run_mamba(ws, ws + OFF_F, DM, LC, 1, m_in_w, DM, m_conv_w, m_conv_b, m_xproj,
            m_dt_w, m_dt_b, m_A_log, m_D, m_out_w, ws + OFF_OUT, stream);
  softmax_rows_k<<<LC, 256, 0, stream>>>(ws + OFF_OUT, ws + OFF_YI, DM);

  // build seq_fi / seq_if  [2049,1024]
  assemble_k<<<cdiv((LC + 1) * 1024, 256), 256, 0, stream>>>(
      ws + OFF_YF, ws + OFF_YI, cls1, cls2, ws + OFF_SFI, ws + OFF_SIF);

  // cross level on seq_fi -> keep last row
  run_mamba(ws, ws + OFF_SFI, 1024, LC + 1, 0, c_in_w, 1024, c_conv_w, c_conv_b,
            c_xproj, c_dt_w, c_dt_b, c_A_log, c_D, c_out_w, ws + OFF_OUT, stream);
  copy_row_k<<<2, 256, 0, stream>>>(ws + OFF_OUT + (long)LC * DM, ws + OFF_LAST);

  // cross level on seq_if -> keep last row
  run_mamba(ws, ws + OFF_SIF, 1024, LC + 1, 0, c_in_w, 1024, c_conv_w, c_conv_b,
            c_xproj, c_dt_w, c_dt_b, c_A_log, c_D, c_out_w, ws + OFF_OUT, stream);
  copy_row_k<<<2, 256, 0, stream>>>(ws + OFF_OUT + (long)LC * DM, ws + OFF_LAST + DM);

  // classifier: [1024] @ [1024,2] + bias
  classifier_k<<<1, 256, 0, stream>>>(ws + OFF_LAST, cls_w, cls_b, out);
}
